// TransformerConvNet_27023934226709
// MI455X (gfx1250) — matrix operand order repaired and checked by reading, not compiled, not run on hardware
//
#include <hip/hip_runtime.h>
#include <hip/hip_bf16.h>

typedef _Float16 half16 __attribute__((ext_vector_type(16)));
typedef float    float8 __attribute__((ext_vector_type(8)));

// ---------- WMMA lane-layout helpers (ISA 7.12.2) ----------

// A-matrix (16x32 f16): half h, lane l -> k = (h>=8?16:0) + (l>=16?8:0) + (h&7)
__device__ __forceinline__ int koffA(int h, int lane) {
  return ((h >> 3) << 4) + ((lane & 16) ? 8 : 0) + (h & 7);
}
// inverse of koffA: given k (0..31) and row m (0..15) -> (lane, h)
__device__ __forceinline__ void invA(int kk, int m, int& lane, int& h) {
  h    = (((kk >> 4) & 1) << 3) + (kk & 7);
  lane = m + (((kk >> 3) & 1) << 4);
}
// B-matrix (32x16 f16): col n = lane%16; same k order as A:
// k = (h>=8?16:0) + (l>=16?8:0) + (h&7).  inverse: given k (0..31), col n -> (lane, h)
__device__ __forceinline__ void invB(int kk, int n, int& lane, int& h) {
  h    = (((kk >> 4) & 1) << 3) + (kk & 7);
  lane = (n & 15) + (((kk >> 3) & 1) << 4);
}

__device__ __forceinline__ void atomicAddF(float* p, float v) {
  __hip_atomic_fetch_add(p, v, __ATOMIC_RELAXED, __HIP_MEMORY_SCOPE_AGENT);
}

__device__ __forceinline__ unsigned orderKey(float f) {
  unsigned b = __float_as_uint(f);
  return (b & 0x80000000u) ? ~b : (b | 0x80000000u);
}
__device__ __forceinline__ float orderDecode(unsigned u) {
  unsigned b = (u & 0x80000000u) ? (u & 0x7FFFFFFFu) : ~u;
  return __uint_as_float(b);
}

__global__ void zero_kernel(unsigned* __restrict__ p, long n) {
  long i = (long)blockIdx.x * 256 + threadIdx.x;
  if (i < n) p[i] = 0u;
}

// ---------- Stage 1: edge MLP (EmbedConv) ----------
// 128 edges per block (8 waves x 16 rows). Two WMMA GEMMs + ReLU + LN, then
// float-atomic scatter-add into xe[dst]. LDS operands are stored pre-swizzled
// in WMMA fragment layout so fragment loads are contiguous ds_load_b128s.
__global__ __launch_bounds__(256) void embed_edge_kernel(
    const float* __restrict__ x, const int* __restrict__ edge_index,
    const float* __restrict__ edge_attr,
    const float* __restrict__ w1, const float* __restrict__ b1,
    const float* __restrict__ g1, const float* __restrict__ be1,
    const float* __restrict__ w2, const float* __restrict__ b2,
    const float* __restrict__ g2, const float* __restrict__ be2,
    float* __restrict__ xe, int nE)
{
  // B-fragment layout: [tile(4) x frag(KF) x lane(32)] of half16
  __shared__ half16 sW1V[4 * 1 * 32];        // w1 rows 0..31, KF=1
  __shared__ half16 sW2V[4 * 2 * 32];        // w2, KF=2
  __shared__ half16 sHnV[8 * 2 * 32];        // LN1 output in A-fragment layout
  __shared__ float  sH [128 * 64];
  __shared__ float  sEA[128];
  __shared__ int    sDst[128];

  _Float16* sW1h = (_Float16*)sW1V;
  _Float16* sW2h = (_Float16*)sW2V;
  _Float16* sHnh = (_Float16*)sHnV;

  const int tid  = threadIdx.x;
  const int wave = tid >> 5, lane = tid & 31, laneN = lane & 15;
  const long eb  = (long)blockIdx.x * 128;

  // stage w1 rows 0..31 (swizzled to B-fragment layout)
  for (int i = tid; i < 32 * 64; i += 256) {
    int kg = i >> 6, n = i & 63;
    int bl, bh; invB(kg, n, bl, bh);
    sW1h[(((n >> 4)) * 32 + bl) * 16 + bh] = (_Float16)w1[i];
  }
  // stage w2 (swizzled)
  for (int i = tid; i < 64 * 64; i += 256) {
    int kg = i >> 6, n = i & 63;
    int f = kg >> 5, kk = kg & 31;
    int bl, bh; invB(kk, n, bl, bh);
    sW2h[(((n >> 4) * 2 + f) * 32 + bl) * 16 + bh] = (_Float16)w2[i];
  }
  if (tid < 128) {
    long e = eb + tid;
    sEA[tid]  = (e < nE) ? edge_attr[e] : 0.0f;
    sDst[tid] = (e < nE) ? edge_index[nE + e] : 0;
  }
  __syncthreads();

  // --- GEMM1: [16 edges x 32] @ w1[0:32,0:64]  (row 32 added as rank-1) ---
  const int row = wave * 16 + laneN;      // block-local edge row for A
  const long e  = eb + row;
  const int src = (e < nE) ? edge_index[e] : 0;
  half16 a;
  for (int h = 0; h < 16; ++h) {
    int kk = koffA(h, lane);              // 0..31
    a[h] = (e < nE) ? (_Float16)x[(long)src * 32 + kk] : (_Float16)0.0f;
  }
#pragma unroll
  for (int t = 0; t < 4; ++t) {
    const half16 b = sW1V[t * 32 + lane];
    float8 acc = {};
    acc = __builtin_amdgcn_wmma_f32_16x16x32_f16(false, a, false, b,
                                                 (short)0, acc, false, false);
    const int c = t * 16 + laneN;
    const float bb = b1[c], w33 = w1[32 * 64 + c];
#pragma unroll
    for (int j = 0; j < 8; ++j) {
      int rr = wave * 16 + j + ((lane & 16) ? 8 : 0);
      float val = acc[j] + bb + sEA[rr] * w33;   // rank-1: edge_attr column
      sH[rr * 64 + c] = fmaxf(val, 0.0f);        // ReLU
    }
  }
  __syncthreads();

  // --- LayerNorm 1 (per edge row), write f16 in A-fragment layout ---
  if (tid < 128) {
    float mu = 0.0f;
    for (int c = 0; c < 64; ++c) mu += sH[tid * 64 + c];
    mu *= (1.0f / 64.0f);
    float var = 0.0f;
    for (int c = 0; c < 64; ++c) { float d = sH[tid * 64 + c] - mu; var += d * d; }
    var *= (1.0f / 64.0f);
    const float rs = rsqrtf(var + 1e-5f);
    const int w = tid >> 4, m = tid & 15;
    for (int c = 0; c < 64; ++c) {
      float v = (sH[tid * 64 + c] - mu) * rs * g1[c] + be1[c];
      int f = c >> 5, kk = c & 31;
      int al, ah; invA(kk, m, al, ah);
      sHnh[((w * 2 + f) * 32 + al) * 16 + ah] = (_Float16)v;
    }
  }
  __syncthreads();

  // --- GEMM2: [16 x 64] @ w2[64 x 64] ---
  half16 a2[2];
#pragma unroll
  for (int f = 0; f < 2; ++f) a2[f] = sHnV[(wave * 2 + f) * 32 + lane];
#pragma unroll
  for (int t = 0; t < 4; ++t) {
    float8 acc = {};
#pragma unroll
    for (int f = 0; f < 2; ++f) {
      const half16 b = sW2V[(t * 2 + f) * 32 + lane];
      acc = __builtin_amdgcn_wmma_f32_16x16x32_f16(false, a2[f], false, b,
                                                   (short)0, acc, false, false);
    }
    const int c = t * 16 + laneN;
    const float bb = b2[c];
#pragma unroll
    for (int j = 0; j < 8; ++j) {
      int rr = wave * 16 + j + ((lane & 16) ? 8 : 0);
      sH[rr * 64 + c] = fmaxf(acc[j] + bb, 0.0f);  // ReLU
    }
  }
  __syncthreads();

  // --- LayerNorm 2 (in place, f32) ---
  if (tid < 128) {
    float mu = 0.0f;
    for (int c = 0; c < 64; ++c) mu += sH[tid * 64 + c];
    mu *= (1.0f / 64.0f);
    float var = 0.0f;
    for (int c = 0; c < 64; ++c) { float d = sH[tid * 64 + c] - mu; var += d * d; }
    var *= (1.0f / 64.0f);
    const float rs = rsqrtf(var + 1e-5f);
    for (int c = 0; c < 64; ++c)
      sH[tid * 64 + c] = (sH[tid * 64 + c] - mu) * rs * g2[c] + be2[c];
  }
  __syncthreads();

  // --- scatter-add to xe[dst] ---
  for (int idx = tid; idx < 128 * 64; idx += 256) {
    int r = idx >> 6, c = idx & 63;
    long ee = eb + r;
    if (ee < nE) atomicAddF(&xe[(long)sDst[r] * 64 + c], sH[idx]);
  }
}

// ---------- Stage 2a: node GEMM -> q,k,v,skip ----------
template <int FI>
__global__ __launch_bounds__(256) void node_qkvs_kernel(
    const float* __restrict__ X,
    const float* __restrict__ wq, const float* __restrict__ bq,
    const float* __restrict__ wk, const float* __restrict__ bk,
    const float* __restrict__ wv, const float* __restrict__ bv,
    const float* __restrict__ wsk, const float* __restrict__ bsk,
    float* __restrict__ q, float* __restrict__ k, float* __restrict__ v,
    float* __restrict__ skip, int N)
{
  constexpr int KF = FI / 32;
  // weight staged per-matrix in B-fragment layout: [tile(8) x frag(KF) x lane(32)]
  __shared__ half16 sWV[8 * KF * 32];
  _Float16* sWh = (_Float16*)sWV;

  const int tid  = threadIdx.x;
  const int wave = tid >> 5, lane = tid & 31, laneN = lane & 15;
  const int r0   = blockIdx.x * 128 + wave * 16;
  const int m    = r0 + laneN;

  half16 a[KF];
#pragma unroll
  for (int f = 0; f < KF; ++f)
    for (int h = 0; h < 16; ++h) {
      int kk = f * 32 + koffA(h, lane);
      a[f][h] = (m < N) ? (_Float16)X[(long)m * FI + kk] : (_Float16)0.0f;
    }

  const float* Ws[4] = {wq, wk, wv, wsk};
  const float* Bs[4] = {bq, bk, bv, bsk};
  float*       Os[4] = {q, k, v, skip};

#pragma unroll
  for (int mat = 0; mat < 4; ++mat) {
    __syncthreads();
    for (int i = tid; i < FI * 128; i += 256) {
      int kg = i >> 7, n = i & 127;
      int f = kg >> 5, kk = kg & 31;
      int bl, bh; invB(kk, n, bl, bh);
      sWh[(((n >> 4) * KF + f) * 32 + bl) * 16 + bh] = (_Float16)Ws[mat][i];
    }
    __syncthreads();
#pragma unroll
    for (int t = 0; t < 8; ++t) {
      float8 acc = {};
#pragma unroll
      for (int f = 0; f < KF; ++f) {
        const half16 b = sWV[(t * KF + f) * 32 + lane];
        acc = __builtin_amdgcn_wmma_f32_16x16x32_f16(false, a[f], false, b,
                                                     (short)0, acc, false, false);
      }
      const int c = t * 16 + laneN;
      const float bias = Bs[mat][c];
#pragma unroll
      for (int j = 0; j < 8; ++j) {
        int mm = r0 + j + ((lane & 16) ? 8 : 0);
        if (mm < N) Os[mat][(long)mm * 128 + c] = acc[j] + bias;
      }
    }
  }
}

// ---------- Stage 2b: attention logits + segment max ----------
__global__ void edge_alpha_kernel(
    const float* __restrict__ q, const float* __restrict__ k,
    const float* __restrict__ we, const int* __restrict__ edge_index,
    const float* __restrict__ edge_attr, float* __restrict__ alpha_raw,
    unsigned* __restrict__ mkey, int nE)
{
  long idx = (long)blockIdx.x * 256 + threadIdx.x;
  if (idx >= (long)nE * 4) return;
  long e = idx >> 2; int h = idx & 3;
  int s = edge_index[e], d = edge_index[nE + e];
  float ea = edge_attr[e];
  const float* qp = q + (long)d * 128 + h * 32;
  const float* kp = k + (long)s * 128 + h * 32;
  const float* wp = we + h * 32;
  float sum = 0.0f;
  for (int i = 0; i < 32; ++i) sum += qp[i] * (kp[i] + ea * wp[i]);
  float alpha = sum * 0.17677669529663687f;  // 1/sqrt(32)
  alpha_raw[idx] = alpha;
  atomicMax(&mkey[(long)d * 4 + h], orderKey(alpha));
}

// ---------- Stage 2c: exp + denom + unnormalized message accumulate ----------
__global__ void edge_msg_kernel(
    const float* __restrict__ v, const float* __restrict__ we,
    const int* __restrict__ edge_index, const float* __restrict__ edge_attr,
    const float* __restrict__ alpha_raw, const unsigned* __restrict__ mkey,
    float* __restrict__ dsum, float* __restrict__ msg, int nE)
{
  long idx = (long)blockIdx.x * 256 + threadIdx.x;
  if (idx >= (long)nE * 4) return;
  long e = idx >> 2; int h = idx & 3;
  int s = edge_index[e], d = edge_index[nE + e];
  float ea = edge_attr[e];
  float m  = orderDecode(mkey[(long)d * 4 + h]);
  float a  = __expf(alpha_raw[idx] - m);
  atomicAddF(&dsum[(long)d * 4 + h], a);
  const float* vp = v + (long)s * 128 + h * 32;
  const float* wp = we + h * 32;
  float* mp = msg + (long)d * 128 + h * 32;
  for (int i = 0; i < 32; ++i) atomicAddF(&mp[i], a * (vp[i] + ea * wp[i]));
}

// ---------- Stage 2d: normalize + skip + ReLU (in place into msg) ----------
__global__ void node_finalize_kernel(float* __restrict__ msg,
                                     const float* __restrict__ dsum,
                                     const float* __restrict__ skip, int N)
{
  long idx = (long)blockIdx.x * 256 + threadIdx.x;
  if (idx >= (long)N * 128) return;
  long n = idx >> 7; int c = idx & 127; int h = c >> 5;
  float val = msg[idx] / (dsum[n * 4 + h] + 1e-16f) + skip[idx];
  msg[idx] = fmaxf(val, 0.0f);
}

// ---------- Stage 3: global mean pool ----------
__global__ void pool_accum_kernel(const float* __restrict__ hbuf,
                                  const int* __restrict__ batch,
                                  float* __restrict__ pooled,
                                  float* __restrict__ cnt, int N)
{
  long idx = (long)blockIdx.x * 256 + threadIdx.x;
  if (idx >= (long)N * 128) return;
  long n = idx >> 7; int c = idx & 127;
  int g = batch[n];
  atomicAddF(&pooled[(long)g * 128 + c], hbuf[idx]);
  if (c == 0) atomicAddF(&cnt[g], 1.0f);
}

__global__ void pool_final_kernel(const float* __restrict__ pooled,
                                  const float* __restrict__ cnt,
                                  float* __restrict__ out)
{
  int idx = blockIdx.x * 256 + threadIdx.x;
  if (idx >= 50 * 128) return;
  int g = idx >> 7;
  out[idx] = pooled[idx] / fmaxf(cnt[g], 1.0f);
}

// ---------- host launcher ----------
extern "C" void kernel_launch(void* const* d_in, const int* in_sizes, int n_in,
                              void* d_out, int out_size, void* d_ws, size_t ws_size,
                              hipStream_t stream)
{
  const float* x          = (const float*)d_in[0];
  const int*   edge_index = (const int*)  d_in[1];
  const float* edge_attr  = (const float*)d_in[2];
  const int*   batch      = (const int*)  d_in[3];
  const float* ew1 = (const float*)d_in[4];
  const float* eb1 = (const float*)d_in[5];
  const float* eg1 = (const float*)d_in[6];
  const float* ebe1= (const float*)d_in[7];
  const float* ew2 = (const float*)d_in[8];
  const float* eb2 = (const float*)d_in[9];
  const float* eg2 = (const float*)d_in[10];
  const float* ebe2= (const float*)d_in[11];
  // layer 0: wq,bq,wk,bk,wv,bv,we,ws,bs starting at 12; layer 1 at 21
  const float* L0[9]; for (int i = 0; i < 9; ++i) L0[i] = (const float*)d_in[12 + i];
  const float* L1[9]; for (int i = 0; i < 9; ++i) L1[i] = (const float*)d_in[21 + i];

  const int N = in_sizes[0] / 32;   // 50000
  const int E = in_sizes[2];        // 500000 (edge_attr flat count, EDGE_DIM=1)

  float* ws = (float*)d_ws;
  size_t off = 0;
  float* xe    = ws + off; off += (size_t)N * 64;
  float* q     = ws + off; off += (size_t)N * 128;
  float* k     = ws + off; off += (size_t)N * 128;
  float* v     = ws + off; off += (size_t)N * 128;
  float* skip  = ws + off; off += (size_t)N * 128;
  float* msg   = ws + off; off += (size_t)N * 128;
  float* dsum  = ws + off; off += (size_t)N * 4;
  float* alpha = ws + off; off += (size_t)E * 4;
  float* pooled= ws + off; off += 50 * 128;
  float* cnt   = ws + off; off += 64;
  unsigned* mkey = (unsigned*)(ws + off); off += (size_t)N * 4;

  auto Z = [&](void* p, long n) {
    zero_kernel<<<(int)((n + 255) / 256), 256, 0, stream>>>((unsigned*)p, n);
  };

  // EmbedConv
  Z(xe, (long)N * 64);
  embed_edge_kernel<<<(E + 127) / 128, 256, 0, stream>>>(
      x, edge_index, edge_attr, ew1, eb1, eg1, ebe1, ew2, eb2, eg2, ebe2, xe, E);

  const long eThreads = (long)E * 4, nThreads = (long)N * 128;
  const int  eBlocks = (int)((eThreads + 255) / 256);
  const int  nBlocks = (int)((nThreads + 255) / 256);

  // TransformerConv layer 0 (fi = 64)
  node_qkvs_kernel<64><<<(N + 127) / 128, 256, 0, stream>>>(
      xe, L0[0], L0[1], L0[2], L0[3], L0[4], L0[5], L0[7], L0[8],
      q, k, v, skip, N);
  Z(mkey, (long)N * 4); Z(dsum, (long)N * 4); Z(msg, (long)N * 128);
  edge_alpha_kernel<<<eBlocks, 256, 0, stream>>>(q, k, L0[6], edge_index,
                                                 edge_attr, alpha, mkey, E);
  edge_msg_kernel<<<eBlocks, 256, 0, stream>>>(v, L0[6], edge_index, edge_attr,
                                               alpha, mkey, dsum, msg, E);
  node_finalize_kernel<<<nBlocks, 256, 0, stream>>>(msg, dsum, skip, N);

  // TransformerConv layer 1 (fi = 128); input h lives in msg
  node_qkvs_kernel<128><<<(N + 127) / 128, 256, 0, stream>>>(
      msg, L1[0], L1[1], L1[2], L1[3], L1[4], L1[5], L1[7], L1[8],
      q, k, v, skip, N);
  Z(mkey, (long)N * 4); Z(dsum, (long)N * 4); Z(msg, (long)N * 128);
  edge_alpha_kernel<<<eBlocks, 256, 0, stream>>>(q, k, L1[6], edge_index,
                                                 edge_attr, alpha, mkey, E);
  edge_msg_kernel<<<eBlocks, 256, 0, stream>>>(v, L1[6], edge_index, edge_attr,
                                               alpha, mkey, dsum, msg, E);
  node_finalize_kernel<<<nBlocks, 256, 0, stream>>>(msg, dsum, skip, N);

  // global mean pool
  Z(pooled, 50 * 128 + 64);   // pooled + cnt are contiguous
  pool_accum_kernel<<<nBlocks, 256, 0, stream>>>(msg, batch, pooled, cnt, N);
  pool_final_kernel<<<(50 * 128 + 255) / 256, 256, 0, stream>>>(pooled, cnt,
                                                                (float*)d_out);
}
